// SocialLSTM_Downsample_53420803228045
// MI455X (gfx1250) — matrix operand order repaired and checked by reading, not compiled, not run on hardware
//
#include <hip/hip_runtime.h>
#include <hip/hip_bf16.h>

// ---------------- problem constants ----------------
#define SEQ   32
#define BATCH 128
#define CELLS 64
#define EMB   64        // E
#define HID   128       // H
#define G4    512       // 4*H
#define CH    8192      // CELLS*HID

typedef __attribute__((ext_vector_type(16))) __bf16 v16bf;
typedef __attribute__((ext_vector_type(8)))  float  v8f;

union BfTile { v16bf v; uint4 q[2]; };

__device__ __forceinline__ unsigned short f2bf(float f) {
  unsigned int u = __builtin_bit_cast(unsigned int, f);
  u += 0x7FFFu + ((u >> 16) & 1u);          // round-to-nearest-even
  return (unsigned short)(u >> 16);
}
__device__ __forceinline__ float sigm(float x) { return 1.f / (1.f + __expf(-x)); }
__device__ __forceinline__ float tanh_f(float x) {
  x = fminf(15.f, fmaxf(-15.f, x));
  float e = __expf(2.f * x);
  return (e - 1.f) / (e + 1.f);
}
__device__ __forceinline__ v8f zero8() { v8f z = {0.f,0.f,0.f,0.f,0.f,0.f,0.f,0.f}; return z; }

// A tile (16x32 bf16) from row-major memory: lane m=l&15 is the M row,
// kh=l>>4 selects the K half. elems[0..7]=K kh*8.. ; elems[8..15]=K 16+kh*8..
__device__ __forceinline__ v16bf load_a(const unsigned short* row, int kbase, int kh) {
  BfTile t;
  t.q[0] = *(const uint4*)(row + kbase + kh * 8);
  t.q[1] = *(const uint4*)(row + kbase + 16 + kh * 8);
  return t.v;
}
// B tile (32x16 bf16) from LDS, tile-major lane-consumption layout:
// piece p (p=0: K kh*8..+7, p=1: K 16+kh*8..+7) of column n lives at
// tileBase + (p*2+kh)*128 + n*8 (elements). 16-lane n-sweep covers all 64 banks.
__device__ __forceinline__ v16bf load_b_lds(const unsigned short* tileBase, int n, int kh) {
  BfTile t;
  t.q[0] = *(const uint4*)(tileBase + (0 * 2 + kh) * 128 + n * 8);
  t.q[1] = *(const uint4*)(tileBase + (1 * 2 + kh) * 128 + n * 8);
  return t.v;
}
#define WMMA_BF16(A_, B_, C_) \
  __builtin_amdgcn_wmma_f32_16x16x32_bf16(false, (A_), false, (B_), (short)0, (C_), false, false)

// ---------------- prep kernels ----------------
__global__ void cvt_bf16_kernel(const float* __restrict__ src,
                                unsigned short* __restrict__ dst, long n) {
  long i = (long)blockIdx.x * blockDim.x + threadIdx.x;
  long stride = (long)gridDim.x * blockDim.x;
  for (; i < n; i += stride) dst[i] = f2bf(src[i]);
}
// soc_W [CH][EMB] -> socWT [EMB][CH]
__global__ void tr_socw_kernel(const float* __restrict__ w,
                               unsigned short* __restrict__ wt) {
  long n = (long)CH * EMB;
  long i = (long)blockIdx.x * blockDim.x + threadIdx.x;
  long stride = (long)gridDim.x * blockDim.x;
  for (; i < n; i += stride) {
    int k = (int)(i >> 6);
    int e = (int)(i & 63);
    wt[(long)e * CH + k] = f2bf(w[i]);
  }
}
// W_ho [C][H][H] (h,o) -> whoT [C][H][H] (o,h)
__global__ void tr_who_kernel(const float* __restrict__ w,
                              unsigned short* __restrict__ wt) {
  long n = (long)CELLS * HID * HID;
  long i = (long)blockIdx.x * blockDim.x + threadIdx.x;
  long stride = (long)gridDim.x * blockDim.x;
  for (; i < n; i += stride) {
    int o = (int)(i & 127);
    int h = (int)((i >> 7) & 127);
    int c = (int)(i >> 14);
    wt[((long)c * HID + o) * HID + h] = f2bf(w[i]);
  }
}
// emb[s,b,c,e] = relu(x[s,b,c,:] @ down_W + down_b), stored bf16
__global__ void emb_kernel(const float* __restrict__ x,
                           const float* __restrict__ dW,   // [3][EMB]
                           const float* __restrict__ db,   // [EMB]
                           unsigned short* __restrict__ embB) {
  long n = (long)SEQ * BATCH * CELLS * EMB;
  long i = (long)blockIdx.x * blockDim.x + threadIdx.x;
  long stride = (long)gridDim.x * blockDim.x;
  for (; i < n; i += stride) {
    int e = (int)(i & 63);
    long sbc = i >> 6;
    const float* xp = x + sbc * 3;
    float v = xp[0] * dW[e] + xp[1] * dW[64 + e] + xp[2] * dW[128 + e] + db[e];
    embB[i] = f2bf(fmaxf(v, 0.f));
  }
}

// ---------------- per-step kernels ----------------
// Split-K social pooling: block p computes partial[p] = h[:, p*512:(p+1)*512] @ socWT^T
// grid = 16 blocks x 256 threads. Weight chunk (64 rows x 512 K = 64KB) staged in LDS.
__global__ void soc_split_kernel(const unsigned short* __restrict__ hbuf,   // [B][CH]
                                 const unsigned short* __restrict__ socWT,  // [EMB][CH]
                                 float* __restrict__ part) {                // [16][B][EMB]
  __shared__ unsigned short smem[64 * 512];   // 64 tiles (kk*4+nt) x 1KB
  const int tid = threadIdx.x;
  const int mt = tid >> 5;
  const int lane = tid & 31;
  const int m = lane & 15;
  const int kh = lane >> 4;
  const int kb0 = blockIdx.x * 512;

  // cooperative stage of the 64KB weight chunk, tile-major lane order
  {
    const int tile = tid >> 2;       // 0..63 : kk*4 + nt
    const int qt   = tid & 3;        // kh2*2 + p
    const int kk = tile >> 2, nt = tile & 3;
    const int kh2 = qt >> 1, p = qt & 1;
    const int kcol = kb0 + kk * 32 + kh2 * 16 + p * 8;
    unsigned short* dst = &smem[tile * 512 + qt * 128];
#pragma unroll
    for (int n = 0; n < 16; ++n) {
      *(uint4*)(dst + n * 8) =
          *(const uint4*)(socWT + (long)(16 * nt + n) * CH + kcol);
    }
  }
  __syncthreads();

  const unsigned short* hRow = hbuf + (long)(16 * mt + m) * CH;
  v8f acc[4];
  acc[0] = zero8(); acc[1] = zero8(); acc[2] = zero8(); acc[3] = zero8();
  for (int kk = 0; kk < 16; ++kk) {
    v16bf At = load_a(hRow, kb0 + kk * 32, kh);
#pragma unroll
    for (int nt = 0; nt < 4; ++nt) {
      v16bf Bt = load_b_lds(&smem[(kk * 4 + nt) * 512], m, kh);
      acc[nt] = WMMA_BF16(At, Bt, acc[nt]);
    }
  }
  float* out = part + (long)blockIdx.x * BATCH * EMB;
#pragma unroll
  for (int nt = 0; nt < 4; ++nt) {
    const int col = 16 * nt + m;
#pragma unroll
    for (int v = 0; v < 8; ++v) {
      const int r = 16 * mt + v + 8 * kh;
      out[(long)r * EMB + col] = acc[nt][v];     // disjoint per block: deterministic
    }
  }
}

// In-order reduction of the 16 partials + bias + relu -> bf16 soc
__global__ void soc_fin_kernel(const float* __restrict__ part,
                               const float* __restrict__ soc_b,
                               unsigned short* __restrict__ soc) {
  const int i = blockIdx.x * blockDim.x + threadIdx.x;   // 0..8191
  if (i >= BATCH * EMB) return;
  float s = 0.f;
#pragma unroll
  for (int p = 0; p < 16; ++p) s += part[(long)p * BATCH * EMB + i];
  s += soc_b[i & 63];
  soc[i] = f2bf(fmaxf(s, 0.f));
}

// Per-cell gate GEMMs + fused LSTM pointwise. grid = CELLS blocks, 256 threads.
// Weights staged through LDS (32KB slice per gate-column group), double-buffered
// so next slice's global loads overlap current slice's WMMAs.
__global__ void cell_kernel(const unsigned short* __restrict__ embB,  // [S][B][C][E]
                            const unsigned short* __restrict__ soc,   // [B][E]
                            unsigned short* __restrict__ hbuf,        // [B][C][H] (in/out)
                            float* __restrict__ cstate,               // [B][C][H]
                            const unsigned short* __restrict__ WihB,  // [C][4H][2E]
                            const unsigned short* __restrict__ WhhB,  // [C][4H][H]
                            const float* __restrict__ b_ih,
                            const float* __restrict__ b_hh,
                            int s) {
  __shared__ unsigned short smem[2][32 * 512];  // 2 x 32 tiles (kt*4+q) x 1KB
  const int c = blockIdx.x;
  const int tid = threadIdx.x;
  const int mt = tid >> 5;
  const int lane = tid & 31;
  const int m = lane & 15;
  const int kh = lane >> 4;
  const int brow = 16 * mt + m;

  const unsigned short* embRow = embB + ((((long)s * BATCH + brow) * CELLS + c) * EMB);
  const unsigned short* socRow = soc + (long)brow * EMB;
  const unsigned short* hRow   = hbuf + ((long)brow * CELLS + c) * HID;

  // hoist all 8 A k-tiles (K = [emb(64) | soc(64) | h(128)])
  v16bf A[8];
  A[0] = load_a(embRow, 0, kh);  A[1] = load_a(embRow, 32, kh);
  A[2] = load_a(socRow, 0, kh);  A[3] = load_a(socRow, 32, kh);
  A[4] = load_a(hRow, 0, kh);    A[5] = load_a(hRow, 32, kh);
  A[6] = load_a(hRow, 64, kh);   A[7] = load_a(hRow, 96, kh);

  const unsigned short* Wih_c = WihB + (long)c * G4 * 128;
  const unsigned short* Whh_c = WhhB + (long)c * G4 * 128;

  // cooperative loader: 256 threads x 128B = one 32KB slice (gate-col group j)
  auto load_slice = [&](int j, int buf) {
    const int unit = tid >> 1;      // 0..127 : 256B quarter of a tile
    const int half = tid & 1;       // which 8 columns
    const int tile = unit >> 2;     // 0..31 : kt*4 + q
    const int qt   = unit & 3;      // kh2*2 + p
    const int kt = tile >> 2;
    const int q  = tile & 3;
    const int kh2 = qt >> 1, p = qt & 1;
    const unsigned short* Wsrc = (kt < 4) ? Wih_c : Whh_c;
    const int kcol = (kt & 3) * 32 + kh2 * 16 + p * 8;
    unsigned short* dst = &smem[buf][tile * 512 + qt * 128 + half * 64];
#pragma unroll
    for (int nn = 0; nn < 8; ++nn) {
      const int n = half * 8 + nn;
      *(uint4*)(dst + nn * 8) =
          *(const uint4*)(Wsrc + (long)(q * HID + 16 * j + n) * 128 + kcol);
    }
  };

  load_slice(0, 0);
  for (int j = 0; j < 8; ++j) {
    __syncthreads();                       // slice j resident in buf j&1
    const int buf = j & 1;
    if (j < 7) load_slice(j + 1, buf ^ 1); // overlap with WMMAs below

    v8f acc[4];
    acc[0] = zero8(); acc[1] = zero8(); acc[2] = zero8(); acc[3] = zero8();
    const unsigned short* sb = smem[buf];
#pragma unroll
    for (int kt = 0; kt < 8; ++kt) {
#pragma unroll
      for (int q = 0; q < 4; ++q) {        // i, f, g, o
        v16bf Bt = load_b_lds(sb + (kt * 4 + q) * 512, m, kh);
        acc[q] = WMMA_BF16(A[kt], Bt, acc[q]);
      }
    }
    // fused LSTM pointwise
    const int col = 16 * j + m;
    const long bb = (long)c * G4;
    const float bi = b_ih[bb + col]           + b_hh[bb + col];
    const float bf = b_ih[bb + HID + col]     + b_hh[bb + HID + col];
    const float bg = b_ih[bb + 2 * HID + col] + b_hh[bb + 2 * HID + col];
    const float bo = b_ih[bb + 3 * HID + col] + b_hh[bb + 3 * HID + col];
#pragma unroll
    for (int v = 0; v < 8; ++v) {
      const int r = 16 * mt + v + 8 * kh;
      const long ci = ((long)r * CELLS + c) * HID + col;
      float iv = sigm(acc[0][v] + bi);
      float fv = sigm(acc[1][v] + bf);
      float gv = tanh_f(acc[2][v] + bg);
      float ov = sigm(acc[3][v] + bo);
      float cn = fv * cstate[ci] + iv * gv;
      float hn = ov * tanh_f(cn);
      cstate[ci] = cn;
      hbuf[ci] = f2bf(hn);
    }
  }
}

// out = sigmoid(h_new @ W_ho + b_ho); y[s,b,c] = out . out_W + out_b
// whoT[c] (32KB) staged once in LDS for the whole block.
__global__ void proj_kernel(const unsigned short* __restrict__ hbuf,   // [B][C][H]
                            const unsigned short* __restrict__ whoT,   // [C][H(o)][H(h)]
                            const float* __restrict__ b_ho,
                            const float* __restrict__ out_W,
                            const float* __restrict__ out_b,
                            float* __restrict__ y,                     // [S][B][C]
                            int s) {
  __shared__ unsigned short smem[32 * 512];   // 32 tiles (j*4+kt) x 1KB
  const int c = blockIdx.x;
  const int tid = threadIdx.x;
  const int mt = tid >> 5;
  const int lane = tid & 31;
  const int m = lane & 15;
  const int kh = lane >> 4;

  const unsigned short* who_c = whoT + (long)c * HID * HID;
  {
    const int unit = tid >> 1;
    const int half = tid & 1;
    const int tile = unit >> 2;     // 0..31 : jj*4 + kt
    const int qt   = unit & 3;
    const int jj = tile >> 2;
    const int kt = tile & 3;
    const int kh2 = qt >> 1, p = qt & 1;
    const int kcol = kt * 32 + kh2 * 16 + p * 8;
    unsigned short* dst = &smem[tile * 512 + qt * 128 + half * 64];
#pragma unroll
    for (int nn = 0; nn < 8; ++nn) {
      const int n = half * 8 + nn;
      *(uint4*)(dst + nn * 8) =
          *(const uint4*)(who_c + (long)(16 * jj + n) * HID + kcol);
    }
  }

  const unsigned short* hRow = hbuf + ((long)(16 * mt + m) * CELLS + c) * HID;
  v16bf A[4];
  A[0] = load_a(hRow, 0, kh);  A[1] = load_a(hRow, 32, kh);
  A[2] = load_a(hRow, 64, kh); A[3] = load_a(hRow, 96, kh);

  __syncthreads();

  float ys[8] = {0.f,0.f,0.f,0.f,0.f,0.f,0.f,0.f};
  for (int j = 0; j < 8; ++j) {
    v8f acc = zero8();
#pragma unroll
    for (int kt = 0; kt < 4; ++kt) {
      v16bf Bt = load_b_lds(&smem[(j * 4 + kt) * 512], m, kh);
      acc = WMMA_BF16(A[kt], Bt, acc);
    }
    const float bho = b_ho[(long)c * HID + 16 * j + m];
    const float ow  = out_W[16 * j + m];
#pragma unroll
    for (int v = 0; v < 8; ++v) ys[v] += sigm(acc[v] + bho) * ow;
  }
  // reduce over the 16 N-lanes sharing each output row
#pragma unroll
  for (int v = 0; v < 8; ++v) {
    float t = ys[v];
    t += __shfl_xor(t, 1, 32);
    t += __shfl_xor(t, 2, 32);
    t += __shfl_xor(t, 4, 32);
    t += __shfl_xor(t, 8, 32);
    if (m == 0) {
      const int r = 16 * mt + v + 8 * kh;
      y[((long)s * BATCH + r) * CELLS + c] = t + out_b[0];
    }
  }
}

// ---------------- host launch ----------------
extern "C" void kernel_launch(void* const* d_in, const int* in_sizes, int n_in,
                              void* d_out, int out_size, void* d_ws, size_t ws_size,
                              hipStream_t stream) {
  const float* x      = (const float*)d_in[0];
  const float* W_ih   = (const float*)d_in[1];
  const float* b_ih   = (const float*)d_in[2];
  const float* W_hh   = (const float*)d_in[3];
  const float* b_hh   = (const float*)d_in[4];
  const float* W_ho   = (const float*)d_in[5];
  const float* b_ho   = (const float*)d_in[6];
  const float* down_W = (const float*)d_in[7];
  const float* down_b = (const float*)d_in[8];
  const float* soc_W  = (const float*)d_in[9];
  const float* soc_b  = (const float*)d_in[10];
  const float* out_W  = (const float*)d_in[11];
  const float* out_b  = (const float*)d_in[12];
  float* y = (float*)d_out;

  char* ws = (char*)d_ws;
  size_t off = 0;
  auto alloc = [&](size_t bytes) -> void* {
    void* p = ws + off;
    off = (off + bytes + 255) & ~(size_t)255;
    return p;
  };
  const long nWih = (long)CELLS * G4 * 128;          // 4,194,304
  const long nEmb = (long)SEQ * BATCH * CELLS * EMB; // 16,777,216
  unsigned short* embB  = (unsigned short*)alloc((size_t)nEmb * 2);
  unsigned short* WihB  = (unsigned short*)alloc((size_t)nWih * 2);
  unsigned short* WhhB  = (unsigned short*)alloc((size_t)nWih * 2);
  unsigned short* socWT = (unsigned short*)alloc((size_t)EMB * CH * 2);
  unsigned short* whoT  = (unsigned short*)alloc((size_t)CELLS * HID * HID * 2);
  unsigned short* hbuf  = (unsigned short*)alloc((size_t)BATCH * CH * 2);
  float*          cst   = (float*)alloc((size_t)BATCH * CH * 4);
  unsigned short* soc   = (unsigned short*)alloc((size_t)BATCH * EMB * 2);
  float*          socP  = (float*)alloc((size_t)16 * BATCH * EMB * 4);
  if (off > ws_size) return;  // workspace too small; bail deterministically

  // state init (graph-capture safe)
  hipMemsetAsync(hbuf, 0, (size_t)BATCH * CH * 2, stream);
  hipMemsetAsync(cst,  0, (size_t)BATCH * CH * 4, stream);

  // prep
  cvt_bf16_kernel<<<(int)((nWih + 255) / 256), 256, 0, stream>>>(W_ih, WihB, nWih);
  cvt_bf16_kernel<<<(int)((nWih + 255) / 256), 256, 0, stream>>>(W_hh, WhhB, nWih);
  tr_socw_kernel<<<(CH * EMB + 255) / 256, 256, 0, stream>>>(soc_W, socWT);
  tr_who_kernel<<<(CELLS * HID * HID + 255) / 256, 256, 0, stream>>>(W_ho, whoT);
  emb_kernel<<<(int)((nEmb + 255) / 256), 256, 0, stream>>>(x, down_W, down_b, embB);

  // recurrence
  for (int s = 0; s < SEQ; ++s) {
    soc_split_kernel<<<16, 256, 0, stream>>>(hbuf, socWT, socP);
    soc_fin_kernel<<<(BATCH * EMB + 255) / 256, 256, 0, stream>>>(socP, soc_b, soc);
    cell_kernel<<<CELLS, 256, 0, stream>>>(embB, soc, hbuf, cst, WihB, WhhB,
                                           b_ih, b_hh, s);
    proj_kernel<<<CELLS, 256, 0, stream>>>(hbuf, whoT, b_ho, out_W, out_b, y, s);
  }
}
